// ResponseDecoder_8967891714616
// MI455X (gfx1250) — matrix operand order repaired and checked by reading, not compiled, not run on hardware
//
#include <hip/hip_runtime.h>
#include <math.h>

// ---------------- problem dimensions ----------------
constexpr int NV  = 8000;          // vocab
constexpr int NTZ = 128;           // z time steps
constexpr int NTU = 256;           // u time steps
constexpr int NB  = 64;            // batch
constexpr int NH  = 1024;          // hidden
constexpr int NE  = 512;           // embed
constexpr int NDG = 64;            // degree
constexpr int NGI = NE + 2 * NH + NDG;   // 2624
constexpr int NH3 = 3 * NH;              // 3072
constexpr int NVC = NV + NTZ;            // 8128

// ---------------- WMMA types ----------------
typedef __bf16 bf16_t;
typedef bf16_t v16bf __attribute__((ext_vector_type(16)));
typedef bf16_t v2bf  __attribute__((ext_vector_type(2)));
typedef float  v8f   __attribute__((ext_vector_type(8)));

// ---------------- fragment loaders ----------------
// A fragment 16x32: lane half 0 -> K0..15,
//                  lane half 1 -> K16..31 (contiguous, matches B), row = lane%16.
__device__ __forceinline__ v16bf load_a_f32(const float* X, int ldx, int row0, int k0) {
  const int lane = threadIdx.x & 31;
  const int m = lane & 15, half = lane >> 4;
  const float* p = X + (size_t)(row0 + m) * ldx + k0;
  v16bf a;
#pragma unroll
  for (int i = 0; i < 8; ++i) a[i] = (bf16_t)p[half * 16 + i];
#pragma unroll
  for (int i = 0; i < 8; ++i) a[8 + i] = (bf16_t)p[half * 16 + 8 + i];
  return a;
}
__device__ __forceinline__ v16bf load_a_bf(const bf16_t* X, int ldx, int row0, int k0) {
  const int lane = threadIdx.x & 31;
  const int m = lane & 15, half = lane >> 4;
  const bf16_t* p = X + (size_t)(row0 + m) * ldx + k0;
  v16bf a;
#pragma unroll
  for (int i = 0; i < 8; ++i) a[i] = p[half * 16 + i];        // 16B contiguous
#pragma unroll
  for (int i = 0; i < 8; ++i) a[8 + i] = p[half * 16 + 8 + i];
  return a;
}
// B fragment 32x16, B = W^T with W row-major (N x K): lane n = lane%16 picks W
// row; lane half picks K 0..15 / 16..31 (contiguous run).
__device__ __forceinline__ v16bf load_b_f32(const float* W, int ldw, int n0, int k0) {
  const int lane = threadIdx.x & 31;
  const int n = lane & 15, half = lane >> 4;
  const float* p = W + (size_t)(n0 + n) * ldw + k0 + half * 16;
  v16bf b;
#pragma unroll
  for (int i = 0; i < 16; ++i) b[i] = (bf16_t)p[i];
  return b;
}
__device__ __forceinline__ v16bf load_b_bf(const bf16_t* W, int ldw, int n0, int k0) {
  const int lane = threadIdx.x & 31;
  const int n = lane & 15, half = lane >> 4;
  const bf16_t* p = W + (size_t)(n0 + n) * ldw + k0 + half * 16;
  v16bf b;
#pragma unroll
  for (int i = 0; i < 16; ++i) b[i] = p[i];                  // 32B contiguous
  return b;
}

// ---------------- one-shot f32 -> bf16 conversion ----------------
__global__ void f32_to_bf16_kernel(const float* __restrict__ src,
                                   bf16_t* __restrict__ dst, long npairs) {
  long i = (long)blockIdx.x * blockDim.x + threadIdx.x;
  if (i >= npairs) return;
  float2 v = ((const float2*)src)[i];
  v2bf o;
  o[0] = (bf16_t)v.x;
  o[1] = (bf16_t)v.y;
  ((v2bf*)dst)[i] = o;
}

// ---------------- GEMM: out = X @ W^T + bias, 2x4 register blocking ----------
// One wave owns a 32x64 output tile. All 6 fragments are loaded before the
// 8 back-to-back WMMAs so the B-register WAR stall is paid once per k-step.
// block = 128 (4 waves, consecutive 64-col groups); grid.y = M/32.
__global__ void gemm_bias_kernel(const float* __restrict__ X, int ldx,
                                 const float* __restrict__ W, int ldw,
                                 const float* __restrict__ bias,
                                 float* __restrict__ out, int ldo, int K,
                                 int ngroups) {
  const int wave = threadIdx.x >> 5;
  const int lane = threadIdx.x & 31;
  const int ng = blockIdx.x * 4 + wave;
  if (ng >= ngroups) return;                 // wave-uniform; EXEC stays full
  const int n0 = ng * 64;
  const int row0 = blockIdx.y * 32;
  v8f acc[2][4];
#pragma unroll
  for (int i = 0; i < 2; ++i)
#pragma unroll
    for (int j = 0; j < 4; ++j)
      acc[i][j] = {0.f, 0.f, 0.f, 0.f, 0.f, 0.f, 0.f, 0.f};
  for (int k0 = 0; k0 < K; k0 += 32) {
    v16bf a0 = load_a_f32(X, ldx, row0, k0);
    v16bf a1 = load_a_f32(X, ldx, row0 + 16, k0);
    v16bf b0 = load_b_f32(W, ldw, n0, k0);
    v16bf b1 = load_b_f32(W, ldw, n0 + 16, k0);
    v16bf b2 = load_b_f32(W, ldw, n0 + 32, k0);
    v16bf b3 = load_b_f32(W, ldw, n0 + 48, k0);
    acc[0][0] = __builtin_amdgcn_wmma_f32_16x16x32_bf16(false, a0, false, b0,
                                                        (short)0, acc[0][0], false, false);
    acc[0][1] = __builtin_amdgcn_wmma_f32_16x16x32_bf16(false, a0, false, b1,
                                                        (short)0, acc[0][1], false, false);
    acc[0][2] = __builtin_amdgcn_wmma_f32_16x16x32_bf16(false, a0, false, b2,
                                                        (short)0, acc[0][2], false, false);
    acc[0][3] = __builtin_amdgcn_wmma_f32_16x16x32_bf16(false, a0, false, b3,
                                                        (short)0, acc[0][3], false, false);
    acc[1][0] = __builtin_amdgcn_wmma_f32_16x16x32_bf16(false, a1, false, b0,
                                                        (short)0, acc[1][0], false, false);
    acc[1][1] = __builtin_amdgcn_wmma_f32_16x16x32_bf16(false, a1, false, b1,
                                                        (short)0, acc[1][1], false, false);
    acc[1][2] = __builtin_amdgcn_wmma_f32_16x16x32_bf16(false, a1, false, b2,
                                                        (short)0, acc[1][2], false, false);
    acc[1][3] = __builtin_amdgcn_wmma_f32_16x16x32_bf16(false, a1, false, b3,
                                                        (short)0, acc[1][3], false, false);
  }
  const int half = lane >> 4;
  const int cl = lane & 15;
#pragma unroll
  for (int i = 0; i < 2; ++i)
#pragma unroll
    for (int j = 0; j < 4; ++j) {
      const int col = n0 + 16 * j + cl;
      const float bv = bias ? bias[col] : 0.f;
#pragma unroll
      for (int r = 0; r < 8; ++r)
        out[(size_t)(row0 + 16 * i + 8 * half + r) * ldo + col] = acc[i][j][r] + bv;
    }
}

// ---------------- fused energy -> tanh -> weighted reduction ----------------
// bf16 operands (pre-converted). Rows = (t*B+b); one wave per 32-row tile,
// 2x2 register blocking (4 loads feed 4 WMMAs). scores[t*B+b].
__global__ void enc_score_kernel(const bf16_t* __restrict__ enc,
                                 const bf16_t* __restrict__ W2, int ldw,
                                 const float* __restrict__ rowterm,  // (B,NH)|null
                                 const float* __restrict__ colterm,  // (NH)  |null
                                 const float* __restrict__ vN,       // (NH)  |null
                                 const float* __restrict__ vBN,      // (B,NH)|null
                                 float* __restrict__ scores) {
  const int wave = threadIdx.x >> 5;
  const int lane = threadIdx.x & 31;
  const int row0 = (blockIdx.x * 4 + wave) * 32;
  const int half = lane >> 4;
  const int cl = lane & 15;
  float sacc[2][8];
#pragma unroll
  for (int i = 0; i < 2; ++i)
#pragma unroll
    for (int r = 0; r < 8; ++r) sacc[i][r] = 0.f;
  for (int nt = 0; nt < NH / 32; ++nt) {
    const int n0 = nt * 32;
    v8f cc[2][2];
#pragma unroll
    for (int i = 0; i < 2; ++i)
#pragma unroll
      for (int j = 0; j < 2; ++j)
        cc[i][j] = {0.f, 0.f, 0.f, 0.f, 0.f, 0.f, 0.f, 0.f};
    for (int k0 = 0; k0 < NH; k0 += 32) {
      v16bf a0 = load_a_bf(enc, NH, row0, k0);
      v16bf a1 = load_a_bf(enc, NH, row0 + 16, k0);
      v16bf b0 = load_b_bf(W2, ldw, n0, k0);
      v16bf b1 = load_b_bf(W2, ldw, n0 + 16, k0);
      cc[0][0] = __builtin_amdgcn_wmma_f32_16x16x32_bf16(false, a0, false, b0,
                                                         (short)0, cc[0][0], false, false);
      cc[0][1] = __builtin_amdgcn_wmma_f32_16x16x32_bf16(false, a0, false, b1,
                                                         (short)0, cc[0][1], false, false);
      cc[1][0] = __builtin_amdgcn_wmma_f32_16x16x32_bf16(false, a1, false, b0,
                                                         (short)0, cc[1][0], false, false);
      cc[1][1] = __builtin_amdgcn_wmma_f32_16x16x32_bf16(false, a1, false, b1,
                                                         (short)0, cc[1][1], false, false);
    }
#pragma unroll
    for (int i = 0; i < 2; ++i)
#pragma unroll
      for (int j = 0; j < 2; ++j) {
        const int n = n0 + 16 * j + cl;
        const float ct = colterm ? colterm[n] : 0.f;
#pragma unroll
        for (int r = 0; r < 8; ++r) {
          const int row = row0 + 16 * i + 8 * half + r;
          const int bb = row & (NB - 1);  // row = t*B + b, B=64
          float e = cc[i][j][r] + ct;
          if (rowterm) e += rowterm[(size_t)bb * NH + n];
          e = tanhf(e);
          const float w = vN ? vN[n] : vBN[(size_t)bb * NH + n];
          sacc[i][r] += e * w;
        }
      }
  }
#pragma unroll
  for (int i = 0; i < 2; ++i)
#pragma unroll
    for (int r = 0; r < 8; ++r) {
      float s = sacc[i][r];
      s += __shfl_xor(s, 1, 32);
      s += __shfl_xor(s, 2, 32);
      s += __shfl_xor(s, 4, 32);
      s += __shfl_xor(s, 8, 32);
      if (cl == 0) scores[row0 + 16 * i + 8 * half + r] = s;
    }
}

// ---------------- softmax over t + context = alpha @ enc (bf16 stream) -------
__global__ void softmax_ctx_kernel(const float* __restrict__ scores,  // [t*B+b]
                                   const bf16_t* __restrict__ enc,    // (T,B,H)
                                   float* __restrict__ ctx,           // (B,H)
                                   int T) {
  __shared__ float red[256];
  __shared__ float al[256];
  const int b = blockIdx.x;
  const int tid = threadIdx.x;
  float s = (tid < T) ? scores[tid * NB + b] : -3.0e38f;
  red[tid] = s;
  __syncthreads();
  for (int o = 128; o > 0; o >>= 1) {
    if (tid < o) red[tid] = fmaxf(red[tid], red[tid + o]);
    __syncthreads();
  }
  const float mx = red[0];
  __syncthreads();
  float e = (tid < T) ? expf(s - mx) : 0.f;
  red[tid] = e;
  __syncthreads();
  for (int o = 128; o > 0; o >>= 1) {
    if (tid < o) red[tid] += red[tid + o];
    __syncthreads();
  }
  const float inv = 1.f / red[0];
  al[tid] = e * inv;
  __syncthreads();
  for (int h = tid; h < NH; h += 256) {
    float acc = 0.f;
    for (int t = 0; t < T; ++t)
      acc += al[t] * (float)enc[((size_t)t * NB + b) * NH + h];
    ctx[(size_t)b * NH + h] = acc;
  }
}

// ---------------- gru_in = [embed | u_ctx | z_ctx | degree] ----------------
__global__ void build_gru_in_kernel(const float* __restrict__ emb,
                                    const int* __restrict__ m_t,
                                    const float* __restrict__ uctx,
                                    const float* __restrict__ zctx,
                                    const float* __restrict__ degree,
                                    float* __restrict__ gin) {
  int idx = blockIdx.x * blockDim.x + threadIdx.x;
  if (idx >= NB * NGI) return;
  int b = idx / NGI, j = idx % NGI;
  float v;
  if (j < NE)
    v = emb[(size_t)m_t[b] * NE + j];
  else if (j < NE + NH)
    v = uctx[(size_t)b * NH + (j - NE)];
  else if (j < NE + 2 * NH)
    v = zctx[(size_t)b * NH + (j - NE - NH)];
  else
    v = degree[b * NDG + (j - NE - 2 * NH)];
  gin[idx] = v;
}

// ---------------- GRU gates ----------------
__global__ void gru_gate_kernel(const float* __restrict__ gi,
                                const float* __restrict__ gh,
                                const float* __restrict__ hprev,
                                float* __restrict__ gout,
                                float* __restrict__ out_newh,
                                float* __restrict__ out_gru) {
  int idx = blockIdx.x * blockDim.x + threadIdx.x;  // B*H
  int b = idx / NH, h = idx % NH;
  size_t base = (size_t)b * NH3;
  float ir = gi[base + h], iz = gi[base + NH + h], in = gi[base + 2 * NH + h];
  float hr = gh[base + h], hz = gh[base + NH + h], hn = gh[base + 2 * NH + h];
  float r = 1.f / (1.f + expf(-(ir + hr)));
  float z = 1.f / (1.f + expf(-(iz + hz)));
  float n = tanhf(in + r * hn);
  float o = (1.f - z) * n + z * hprev[idx];
  gout[idx] = o;
  out_newh[idx] = o;
  out_gru[idx] = o;
}

// ---------------- gen_in = [z_ctx | u_ctx | gru_out] ----------------
__global__ void build_gen_in_kernel(const float* __restrict__ zctx,
                                    const float* __restrict__ uctx,
                                    const float* __restrict__ gout,
                                    float* __restrict__ genin) {
  int idx = blockIdx.x * blockDim.x + threadIdx.x;  // B*3H
  int b = idx / NH3, j = idx % NH3;
  float v;
  if (j < NH)
    v = zctx[(size_t)b * NH + j];
  else if (j < 2 * NH)
    v = uctx[(size_t)b * NH + (j - NH)];
  else
    v = gout[(size_t)b * NH + (j - 2 * NH)];
  genin[idx] = v;
}

// ---------------- per-b max + exp over copy scores ----------------
__global__ void zcs_maxexp_kernel(const float* __restrict__ zcs,  // [t*B+b]
                                  float* __restrict__ e, float* __restrict__ m) {
  __shared__ float red[NTZ];
  const int b = blockIdx.x, t = threadIdx.x;  // NTZ threads
  float s = zcs[t * NB + b];
  red[t] = s;
  __syncthreads();
  for (int o = NTZ / 2; o > 0; o >>= 1) {
    if (t < o) red[t] = fmaxf(red[t], red[t + o]);
    __syncthreads();
  }
  const float mx = red[0];
  e[t * NB + b] = expf(s - mx);
  if (t == 0) m[b] = mx;
}

// ---------------- z_copy = log(e @ sparse) + m  (streaming, float4) --------
__global__ void sparse_copy_kernel(const float* __restrict__ e,  // [t*B+b]
                                   const float* __restrict__ m,
                                   const float* __restrict__ sp,  // (B,TZ,VC)
                                   float* __restrict__ zcopy) {   // (B,VC)
  __shared__ float es[NTZ];
  const int b = blockIdx.y;
  const int tid = threadIdx.x;
  if (tid < NTZ) es[tid] = e[tid * NB + b];
  __syncthreads();
  const int q = blockIdx.x * blockDim.x + tid;  // float4 column index
  if (q >= NVC / 4) return;
  const float4* spp = (const float4*)(sp + (size_t)b * NTZ * NVC) + q;
  float4 acc = {0.f, 0.f, 0.f, 0.f};
  for (int t = 0; t < NTZ; ++t) {
    const float w = es[t];
    const float4 v = spp[(size_t)t * (NVC / 4)];
    acc.x += w * v.x;
    acc.y += w * v.y;
    acc.z += w * v.z;
    acc.w += w * v.w;
  }
  const float mb = m[b];
  float4 r;
  r.x = logf(acc.x) + mb;
  r.y = logf(acc.y) + mb;
  r.z = logf(acc.z) + mb;
  r.w = logf(acc.w) + mb;
  ((float4*)(zcopy + (size_t)b * NVC))[q] = r;
}

// ---------------- final softmax over [gen_score | z_copy] -------------------
__global__ void final_softmax_kernel(const float* __restrict__ gen,  // (B,V)
                                     const float* __restrict__ zc,   // (B,VC)
                                     float* __restrict__ out) {      // (B,VC)
  __shared__ float red[256];
  const int b = blockIdx.x, tid = threadIdx.x;
  const float* g = gen + (size_t)b * NV;
  const float* z = zc + (size_t)b * NVC;
  float mx = -3.0e38f;
  for (int i = tid; i < NV + NVC; i += 256) {
    float v = (i < NV) ? g[i] : z[i - NV];
    mx = fmaxf(mx, v);
  }
  red[tid] = mx;
  __syncthreads();
  for (int o = 128; o > 0; o >>= 1) {
    if (tid < o) red[tid] = fmaxf(red[tid], red[tid + o]);
    __syncthreads();
  }
  mx = red[0];
  __syncthreads();
  float sm = 0.f;
  for (int i = tid; i < NV + NVC; i += 256) {
    float v = (i < NV) ? g[i] : z[i - NV];
    sm += expf(v - mx);
  }
  red[tid] = sm;
  __syncthreads();
  for (int o = 128; o > 0; o >>= 1) {
    if (tid < o) red[tid] += red[tid + o];
    __syncthreads();
  }
  const float inv = 1.f / red[0];
  float* op = out + (size_t)b * NVC;
  for (int v = tid; v < NVC; v += 256) {
    float p = expf(z[v] - mx) * inv;
    if (v < NV) p += expf(g[v] - mx) * inv;
    op[v] = p;
  }
}

// ---------------- launcher ----------------
extern "C" void kernel_launch(void* const* d_in, const int* in_sizes, int n_in,
                              void* d_out, int out_size, void* d_ws, size_t ws_size,
                              hipStream_t stream) {
  const float* z_enc  = (const float*)d_in[0];
  const float* u_enc  = (const float*)d_in[1];
  const int*   m_t    = (const int*)d_in[2];
  const float* degree = (const float*)d_in[3];
  const float* hprev  = (const float*)d_in[4];
  const float* sparse = (const float*)d_in[5];
  const float* emb    = (const float*)d_in[6];
  const float* azW    = (const float*)d_in[7];
  const float* azb    = (const float*)d_in[8];
  const float* azv    = (const float*)d_in[9];
  const float* auW    = (const float*)d_in[10];
  const float* aub    = (const float*)d_in[11];
  const float* auv    = (const float*)d_in[12];
  const float* Wih    = (const float*)d_in[13];
  const float* Whh    = (const float*)d_in[14];
  const float* bih    = (const float*)d_in[15];
  const float* bhh    = (const float*)d_in[16];
  const float* pW     = (const float*)d_in[17];
  const float* pb     = (const float*)d_in[18];
  const float* cW     = (const float*)d_in[19];
  const float* cb     = (const float*)d_in[20];

  // ---- f32 scratch (~8.6 MB) ----
  float* ws = (float*)d_ws;
  float* hWz   = ws;  ws += (size_t)NB * NH;
  float* hWu   = ws;  ws += (size_t)NB * NH;
  float* scz   = ws;  ws += (size_t)NB * NTZ;
  float* scu   = ws;  ws += (size_t)NB * NTU;
  float* zctx  = ws;  ws += (size_t)NB * NH;
  float* uctx  = ws;  ws += (size_t)NB * NH;
  float* gin   = ws;  ws += (size_t)NB * NGI;
  float* gi    = ws;  ws += (size_t)NB * NH3;
  float* gh    = ws;  ws += (size_t)NB * NH3;
  float* gout  = ws;  ws += (size_t)NB * NH;
  float* genin = ws;  ws += (size_t)NB * NH3;
  float* gens  = ws;  ws += (size_t)NB * NV;
  float* zcs   = ws;  ws += (size_t)NB * NTZ;
  float* ez    = ws;  ws += (size_t)NB * NTZ;
  float* mz    = ws;  ws += (size_t)NB;
  float* zcopy = ws;  ws += (size_t)NB * NVC;
  // ---- bf16 scratch for heavily reused operands (~61 MB) ----
  bf16_t* bw = (bf16_t*)ws;
  bf16_t* zenc_bf = bw;  bw += (size_t)NTZ * NB * NH;   // 8.39M
  bf16_t* uenc_bf = bw;  bw += (size_t)NTU * NB * NH;   // 16.78M
  bf16_t* azW_bf  = bw;  bw += (size_t)NH * 2 * NH;     // 2.10M
  bf16_t* auW_bf  = bw;  bw += (size_t)NH * 2 * NH;     // 2.10M
  bf16_t* cW_bf   = bw;  bw += (size_t)NH * NH;         // 1.05M

  float* out       = (float*)d_out;
  float* out_proba = out;                          // (B, NVC)
  float* out_newh  = out + (size_t)NB * NVC;       // (1,B,H)
  float* out_gru   = out_newh + (size_t)NB * NH;   // (1,B,H)

  // 0) one-shot bf16 conversion of reused operands (halves their bytes,
  //    removes per-tile cvt from the WMMA hot loops)
  {
    long p;
    p = (long)NTZ * NB * NH / 2;
    f32_to_bf16_kernel<<<(p + 255) / 256, 256, 0, stream>>>(z_enc, zenc_bf, p);
    p = (long)NTU * NB * NH / 2;
    f32_to_bf16_kernel<<<(p + 255) / 256, 256, 0, stream>>>(u_enc, uenc_bf, p);
    p = (long)NH * 2 * NH / 2;
    f32_to_bf16_kernel<<<(p + 255) / 256, 256, 0, stream>>>(azW, azW_bf, p);
    f32_to_bf16_kernel<<<(p + 255) / 256, 256, 0, stream>>>(auW, auW_bf, p);
    p = (long)NH * NH / 2;
    f32_to_bf16_kernel<<<(p + 255) / 256, 256, 0, stream>>>(cW, cW_bf, p);
  }

  // 1) hidden half of attention energy (bias folded in); M=64, convert-on-fly
  gemm_bias_kernel<<<dim3(4, 2), 128, 0, stream>>>(hprev, NH, azW, 2 * NH, azb,
                                                   hWz, NH, NH, 16);
  gemm_bias_kernel<<<dim3(4, 2), 128, 0, stream>>>(hprev, NH, auW, 2 * NH, aub,
                                                   hWu, NH, NH, 16);
  // 2) fused energy->tanh->dot(v) scores, bf16 operands, 2x2 blocked
  enc_score_kernel<<<NTZ * NB / 128, 128, 0, stream>>>(zenc_bf, azW_bf + NH,
                                                       2 * NH, hWz, nullptr, azv,
                                                       nullptr, scz);
  enc_score_kernel<<<NTU * NB / 128, 128, 0, stream>>>(uenc_bf, auW_bf + NH,
                                                       2 * NH, hWu, nullptr, auv,
                                                       nullptr, scu);
  // 3) softmax + context (bf16 encoder stream)
  softmax_ctx_kernel<<<NB, 256, 0, stream>>>(scz, zenc_bf, zctx, NTZ);
  softmax_ctx_kernel<<<NB, 256, 0, stream>>>(scu, uenc_bf, uctx, NTU);
  // 4) GRU
  build_gru_in_kernel<<<(NB * NGI + 255) / 256, 256, 0, stream>>>(emb, m_t, uctx,
                                                                  zctx, degree, gin);
  gemm_bias_kernel<<<dim3(12, 2), 128, 0, stream>>>(gin, NGI, Wih, NGI, bih, gi,
                                                    NH3, NGI, 48);
  gemm_bias_kernel<<<dim3(12, 2), 128, 0, stream>>>(hprev, NH, Whh, NH, bhh, gh,
                                                    NH3, NH, 48);
  gru_gate_kernel<<<NB * NH / 256, 256, 0, stream>>>(gi, gh, hprev, gout,
                                                     out_newh, out_gru);
  // 5) vocab projection
  build_gen_in_kernel<<<NB * NH3 / 256, 256, 0, stream>>>(zctx, uctx, gout, genin);
  gemm_bias_kernel<<<dim3(32, 2), 128, 0, stream>>>(genin, NH3, pW, NH3, pb,
                                                    gens, NV, NH3, 125);
  // 6) copy scores: tanh(enc @ copy2^T + b) . gru_out
  enc_score_kernel<<<NTZ * NB / 128, 128, 0, stream>>>(zenc_bf, cW_bf, NH,
                                                       nullptr, cb, nullptr, gout,
                                                       zcs);
  zcs_maxexp_kernel<<<NB, NTZ, 0, stream>>>(zcs, ez, mz);
  // 7) sparse contraction (bandwidth-bound, 266 MB stream)
  sparse_copy_kernel<<<dim3((NVC / 4 + 255) / 256, NB), 256, 0, stream>>>(
      ez, mz, sparse, zcopy);
  // 8) final softmax + fold copy mass into vocab
  final_softmax_kernel<<<NB, 256, 0, stream>>>(gens, zcopy, out_proba);
}
